// DeltaAttention_88596585382721
// MI455X (gfx1250) — matrix operand order repaired and checked by reading, not compiled, not run on hardware
//
#include <hip/hip_runtime.h>
#include <hip/hip_bf16.h>

// ---------------- problem constants ----------------
#define B_SZ   4
#define L_SZ   4096
#define DMODEL 1024
#define NH     16
#define HD     64          // head dim
#define CH     64          // chunk
#define NCHUNK (L_SZ / CH) // 64

typedef __attribute__((ext_vector_type(16))) _Float16 v16h;
typedef __attribute__((ext_vector_type(8)))  _Float16 v8h;
typedef __attribute__((ext_vector_type(8)))  float    v8f;

#define CAT16(lo, hi) __builtin_shufflevector(lo, hi, 0,1,2,3,4,5,6,7,8,9,10,11,12,13,14,15)

// ---------------- async global->LDS copy (gfx1250) ----------------
#if defined(__HIP_DEVICE_COMPILE__) && __has_builtin(__builtin_amdgcn_global_load_async_to_lds_b128)
#define HAVE_ASYNC_LDS 1
#endif

#if defined(HAVE_ASYNC_LDS)
typedef int async_v4i __attribute__((vector_size(16)));           // matches builtin param type
typedef __attribute__((address_space(1))) async_v4i* g_v4i_ptr;   // global
typedef __attribute__((address_space(3))) async_v4i* l_v4i_ptr;   // LDS
#endif

__device__ __forceinline__ void async_copy16(const void* gsrc, void* lds_dst) {
#if defined(HAVE_ASYNC_LDS)
  __builtin_amdgcn_global_load_async_to_lds_b128(
      (g_v4i_ptr)gsrc, (l_v4i_ptr)lds_dst, 0, 0);
#else
  *(float4*)lds_dst = *(const float4*)gsrc;
#endif
}
__device__ __forceinline__ void async_wait0() {
#if defined(HAVE_ASYNC_LDS)
#if __has_builtin(__builtin_amdgcn_s_wait_asynccnt)
  __builtin_amdgcn_s_wait_asynccnt(0);
#else
  asm volatile("s_wait_asynccnt 0" ::: "memory");
#endif
#endif
}
__device__ __forceinline__ void async_wait1() {
#if defined(HAVE_ASYNC_LDS)
#if __has_builtin(__builtin_amdgcn_s_wait_asynccnt)
  __builtin_amdgcn_s_wait_asynccnt(1);
#else
  asm volatile("s_wait_asynccnt 1" ::: "memory");
#endif
#endif
}

// ---------------- WMMA helper (CDNA5 gfx1250) ----------------
__device__ __forceinline__ v8f wmma_f16(v16h a, v16h b, v8f c) {
  return __builtin_amdgcn_wmma_f32_16x16x32_f16(
      false, a, false, b, (short)0, c, false, false);
}

// ---- fragment loaders per ISA 7.12.2 (wave32), vectorized (2 x b128/lane) ----
// A 16x32 f16 from row-major [16 x ld]: lane row = lane&15; per lane the two
// 8-element K-groups {hi*8..hi*8+7} and {16+hi*8..} are CONTIGUOUS.
__device__ __forceinline__ v16h lds_a_frag(const _Float16* src, int ld, int lane) {
  const _Float16* p = src + (size_t)(lane & 15) * ld + ((lane >> 4) << 3);
  v8h lo = *(const v8h*)p;
  v8h hh = *(const v8h*)(p + 16);
  return CAT16(lo, hh);
}
// A frag where logical A[m][k] = src[k][m] (transposed source, e.g. k^T) — gather
__device__ __forceinline__ v16h lds_a_fragT(const _Float16* src, int ld, int lane) {
  int row = lane & 15, hi = lane >> 4;
  v16h a;
#pragma unroll
  for (int e = 0; e < 16; ++e) {
    int kk = ((e >> 3) << 4) + (hi << 3) + (e & 7);
    a[e] = src[(size_t)kk * ld + row];
  }
  return a;
}
// B 32x16 where logical B[k][n] = src[n][k] (N-major source): lane col = lane&15,
// K-groups {hi*8..hi*8+7} and {16+hi*8..}.
__device__ __forceinline__ v16h lds_b_fragT(const _Float16* src, int ld, int lane) {
  const _Float16* p = src + (size_t)(lane & 15) * ld + ((lane >> 4) << 3);
  v8h lo = *(const v8h*)p;
  v8h hh = *(const v8h*)(p + 16);
  return CAT16(lo, hh);
}
// A 16x32 from row-major f32 LDS tile, converting to f16
__device__ __forceinline__ v16h lds_a_frag_f32(const float* src, int ld, int lane) {
  const float* p = src + (size_t)(lane & 15) * ld + ((lane >> 4) << 3);
  float4 a0 = *(const float4*)p,        a1 = *(const float4*)(p + 4);
  float4 a2 = *(const float4*)(p + 16), a3 = *(const float4*)(p + 20);
  v16h a;
  a[0]=(_Float16)a0.x; a[1]=(_Float16)a0.y; a[2]=(_Float16)a0.z; a[3]=(_Float16)a0.w;
  a[4]=(_Float16)a1.x; a[5]=(_Float16)a1.y; a[6]=(_Float16)a1.z; a[7]=(_Float16)a1.w;
  a[8]=(_Float16)a2.x; a[9]=(_Float16)a2.y; a[10]=(_Float16)a2.z; a[11]=(_Float16)a2.w;
  a[12]=(_Float16)a3.x; a[13]=(_Float16)a3.y; a[14]=(_Float16)a3.z; a[15]=(_Float16)a3.w;
  return a;
}
// B[k][n] = W[n][k] from f32 global (weight rows), K-groups {hi*8..} and {16+hi*8..}
__device__ __forceinline__ v16h glb_b_fragT_f32(const float* src, int ld, int lane) {
  const float* p = src + (size_t)(lane & 15) * ld + ((lane >> 4) << 3);
  float4 b0 = *(const float4*)p,       b1 = *(const float4*)(p + 4);
  float4 b2 = *(const float4*)(p + 16), b3 = *(const float4*)(p + 20);
  v16h b;
  b[0]=(_Float16)b0.x; b[1]=(_Float16)b0.y; b[2]=(_Float16)b0.z; b[3]=(_Float16)b0.w;
  b[4]=(_Float16)b1.x; b[5]=(_Float16)b1.y; b[6]=(_Float16)b1.z; b[7]=(_Float16)b1.w;
  b[8]=(_Float16)b2.x; b[9]=(_Float16)b2.y; b[10]=(_Float16)b2.z; b[11]=(_Float16)b2.w;
  b[12]=(_Float16)b3.x; b[13]=(_Float16)b3.y; b[14]=(_Float16)b3.z; b[15]=(_Float16)b3.w;
  return b;
}

// =====================================================================
// Kernel 1: beta = sigmoid(x @ Wb^T + bb), stored [B, H, L] f32
// =====================================================================
__global__ void beta_kernel(const float* __restrict__ x, const float* __restrict__ Wb,
                            const float* __restrict__ bb, float* __restrict__ beta) {
  int t = blockIdx.x * blockDim.x + threadIdx.x;
  if (t >= B_SZ * L_SZ * NH) return;
  int h = t & (NH - 1);
  int l = (t >> 4) & (L_SZ - 1);
  int b = t >> 16;
  const float4* xp = (const float4*)(x + ((size_t)b * L_SZ + l) * DMODEL);
  const float4* wp = (const float4*)(Wb + (size_t)h * DMODEL);
  float s = bb[h];
#pragma unroll 4
  for (int i = 0; i < DMODEL / 4; ++i) {
    float4 xv = xp[i], wv = wp[i];
    s += xv.x * wv.x + xv.y * wv.y + xv.z * wv.z + xv.w * wv.w;
  }
  beta[((size_t)b * NH + h) * L_SZ + l] = 1.f / (1.f + __expf(-s));
}

// =====================================================================
// Kernel 2: Q/K/V projection GEMM. Block = 128 thr (4 waves), block tile
// 16 x 256: A tile (16x32 f32) async-staged in LDS (double buffered) and
// shared by all waves; each wave owns 4 N-tiles -> 4 WMMAs per A fragment.
// =====================================================================
__global__ void proj_qkv_kernel(const float* __restrict__ x,
                                const float* __restrict__ Wq, const float* __restrict__ bq,
                                const float* __restrict__ Wk, const float* __restrict__ bk,
                                const float* __restrict__ Wv, const float* __restrict__ bv,
                                _Float16* __restrict__ qf, _Float16* __restrict__ kf,
                                _Float16* __restrict__ vf) {
  __shared__ __align__(16) float xs[2][16 * 32];
  int t = threadIdx.x, lane = t & 31, wid = t >> 5;
  int mt  = blockIdx.x;                 // 16-row tile
  int ntb = blockIdx.y;                 // 256-col block
  const float* W; const float* bias; _Float16* dst; float scale;
  if (blockIdx.z == 0)      { W = Wq; bias = bq; dst = qf; scale = 0.125f; }
  else if (blockIdx.z == 1) { W = Wk; bias = bk; dst = kf; scale = 1.f; }
  else                      { W = Wv; bias = bv; dst = vf; scale = 1.f; }

  const float* abase = x + (size_t)mt * 16 * DMODEL;
  auto stage = [&](int buf, int k0) {
    int row = t >> 3, cw = (t & 7) << 2;     // 128 thr x 16B = 2KB tile
    async_copy16(abase + (size_t)row * DMODEL + k0 + cw, &xs[buf][row * 32 + cw]);
  };

  v8f acc[4] = {};
  stage(0, 0);
  int buf = 0;
  for (int k0 = 0; k0 < DMODEL; k0 += 32) {
    bool more = (k0 + 32) < DMODEL;
    if (more) { stage(buf ^ 1, k0 + 32); async_wait1(); } else { async_wait0(); }
    __syncthreads();
    v16h a = lds_a_frag_f32(xs[buf], 32, lane);
    const float* wb = W + (size_t)(ntb * 256 + wid * 64) * DMODEL + k0;
    __builtin_prefetch(wb + (size_t)(lane & 15) * DMODEL + 128, 0, 1);
#pragma unroll
    for (int s3 = 0; s3 < 4; ++s3) {
      v16h b = glb_b_fragT_f32(wb + (size_t)s3 * 16 * DMODEL, DMODEL, lane);
      acc[s3] = wmma_f16(a, b, acc[s3]);
    }
    __syncthreads();
    buf ^= 1;
  }
  int col = lane & 15, hi = lane >> 4;
#pragma unroll
  for (int s3 = 0; s3 < 4; ++s3) {
    int n  = ntb * 256 + wid * 64 + s3 * 16 + col;
    int h  = n >> 6, hd = n & 63;
    float bi = bias[n];
#pragma unroll
    for (int r = 0; r < 8; ++r) {
      int m  = mt * 16 + hi * 8 + r;
      int b_ = m >> 12;
      int l  = m & (L_SZ - 1);
      dst[(((size_t)b_ * NH + h) * L_SZ + l) * HD + hd] = (_Float16)((acc[s3][r] + bi) * scale);
    }
  }
}

// =====================================================================
// Kernel 3: per-chunk prep. k,v async-staged; A = strict_tril(kb k^T) via
// WMMA (kb formed on the fly from k * beta); in-place forward substitution
// gives u = (I+A)^-1 (v*b), w = (I+A)^-1 (k*b). LDS = 64KB, 256 threads.
// =====================================================================
__global__ void chunk_prep_kernel(const _Float16* __restrict__ kf, const _Float16* __restrict__ vf,
                                  const float* __restrict__ beta,
                                  _Float16* __restrict__ uf, _Float16* __restrict__ wf) {
  __shared__ __align__(16) _Float16 k_sh[CH * HD];  // 8KB
  __shared__ __align__(16) _Float16 v_sh[CH * HD];  // 8KB
  __shared__ float A_sh[CH * CH];                   // 16KB
  __shared__ float u_sh[CH * CH];                   // 16KB (init v*b, solved in place)
  __shared__ float w_sh[CH * CH];                   // 16KB (init k*b, solved in place)

  int bid = blockIdx.x;
  int n  = bid & (NCHUNK - 1);
  int bh = bid >> 6;
  size_t base = ((size_t)bh * L_SZ + (size_t)n * CH) * HD;
  const float* betag = beta + (size_t)bh * L_SZ + (size_t)n * CH;

  int t = threadIdx.x, lane = t & 31, wid = t >> 5;
#pragma unroll
  for (int i = 0; i < 2; ++i) {                     // 256 thr x 2 x 16B = 8KB
    int off = (i * 256 + t) * 8;
    async_copy16(kf + base + off, k_sh + off);
    async_copy16(vf + base + off, v_sh + off);
  }
  async_wait0();
  __syncthreads();

  for (int i = t; i < CH * HD; i += 256) {
    float be = betag[i >> 6];
    u_sh[i] = (float)v_sh[i] * be;
    w_sh[i] = (float)k_sh[i] * be;
  }
  __syncthreads();

  // A = strict_tril((k*b) @ k^T), 16 tiles over 8 waves, K=64
  for (int tile = wid; tile < 16; tile += 8) {
    int ti = tile >> 2, tj = tile & 3;
    _Float16 beh = (_Float16)betag[ti * 16 + (lane & 15)];  // per-lane row scale
    v8f acc = {};
    for (int k0 = 0; k0 < HD; k0 += 32) {
      v16h a = lds_a_frag(k_sh + (size_t)ti * 16 * HD + k0, HD, lane) * beh;
      v16h b = lds_b_fragT(k_sh + (size_t)tj * 16 * HD + k0, HD, lane);
      acc = wmma_f16(a, b, acc);
    }
    int col = lane & 15, hi = lane >> 4;
    int gj = tj * 16 + col;
#pragma unroll
    for (int r = 0; r < 8; ++r) {
      int gi = ti * 16 + hi * 8 + r;
      A_sh[gi * CH + gj] = (gi > gj) ? acc[r] : 0.f;
    }
  }
  __syncthreads();

  // Column-per-thread forward substitution (bank-conflict-free, no syncs)
  if (t < 128) {
    int c = t & 63;
    float* dc = (t < 64) ? u_sh : w_sh;
    for (int i = 1; i < CH; ++i) {
      float s = dc[i * CH + c];
      for (int j = 0; j < i; ++j) s -= A_sh[i * CH + j] * dc[j * CH + c];
      dc[i * CH + c] = s;
    }
  }
  __syncthreads();

  for (int i = t; i < CH * HD; i += 256) {
    uf[base + i] = (_Float16)u_sh[i];
    wf[base + i] = (_Float16)w_sh[i];
  }
}

// =====================================================================
// Kernel 4: chunk scan. B*H*4 blocks (value-column split for occupancy),
// 128 thr. q/k/w double-buffered via async-LDS: chunk n+1 streams in while
// chunk n computes. State slice S[64x16] kept f32 + transposed f16 mirror
// (S^T, u^T stored N-major so ALL B-fragments are contiguous b128 loads).
// =====================================================================
__global__ void scan_kernel(const _Float16* __restrict__ qf, const _Float16* __restrict__ kf,
                            const _Float16* __restrict__ uf, const _Float16* __restrict__ wf,
                            _Float16* __restrict__ of) {
  __shared__ __align__(16) _Float16 q_sh[2][CH * HD]; // 16KB
  __shared__ __align__(16) _Float16 k_sh[2][CH * HD]; // 16KB
  __shared__ __align__(16) _Float16 w_sh[2][CH * HD]; // 16KB
  __shared__ __align__(16) _Float16 at_h[CH * CH];    // 8KB  tril(q k^T)
  __shared__ __align__(16) _Float16 S_hT[16 * HD];    // 2KB  S^T slice (N-major)
  __shared__ __align__(16) _Float16 u_hT[16 * CH];    // 2KB  u_i^T slice
  __shared__ float S[HD * 16];                        // 4KB  f32 state slice

  int bid = blockIdx.x;
  int eg  = bid & 3;
  int bh  = bid >> 2;
  int ec  = eg * 16;
  size_t hb = (size_t)bh * L_SZ * HD;

  int t = threadIdx.x, lane = t & 31, wid = t >> 5;
  int col = lane & 15, hi = lane >> 4;

  auto stage = [&](int buf, int n) {
    size_t cb = hb + (size_t)n * CH * HD;
#pragma unroll
    for (int i = 0; i < 4; ++i) {                     // 128 thr x 4 x 16B = 8KB each
      int off = (i * 128 + t) * 8;
      async_copy16(qf + cb + off, &q_sh[buf][off]);
      async_copy16(kf + cb + off, &k_sh[buf][off]);
      async_copy16(wf + cb + off, &w_sh[buf][off]);
    }
  };

  for (int i = t; i < HD * 16; i += 128) S[i] = 0.f;
  for (int i = t; i < 16 * HD; i += 128) S_hT[i] = (_Float16)0.f;
  stage(0, 0);
  async_wait0();
  __syncthreads();

  int buf = 0;
  for (int n = 0; n < NCHUNK; ++n) {
    size_t cb = hb + (size_t)n * CH * HD;
    if (n + 1 < NCHUNK) stage(buf ^ 1, n + 1);        // overlap DMA with compute

    // Phase B: u_i = u0 - w @ S  (one 16x16 tile per wave)
    {
      int ti = wid;
      v8f acc = {};
      for (int k0 = 0; k0 < HD; k0 += 32) {
        v16h a = lds_a_frag(&w_sh[buf][(size_t)ti * 16 * HD + k0], HD, lane);
        v16h b = lds_b_fragT(S_hT + k0, HD, lane);
        acc = wmma_f16(a, b, acc);
      }
#pragma unroll
      for (int r = 0; r < 8; ++r) {
        int row = ti * 16 + hi * 8 + r;
        float u0 = (float)uf[cb + (size_t)row * HD + ec + col];
        u_hT[col * CH + row] = (_Float16)(u0 - acc[r]);
      }
    }
    // attn = incl_tril(q k^T): 16 tiles over 4 waves
    for (int tile = wid; tile < 16; tile += 4) {
      int ti = tile >> 2, tj = tile & 3;
      v8f at = {};
      for (int k0 = 0; k0 < HD; k0 += 32) {
        v16h a = lds_a_frag(&q_sh[buf][(size_t)ti * 16 * HD + k0], HD, lane);
        v16h b = lds_b_fragT(&k_sh[buf][(size_t)tj * 16 * HD + k0], HD, lane);
        at = wmma_f16(a, b, at);
      }
#pragma unroll
      for (int r = 0; r < 8; ++r) {
        int gi = ti * 16 + hi * 8 + r, gj = tj * 16 + col;
        at_h[gi * CH + gj] = (gi >= gj) ? (_Float16)at[r] : (_Float16)0.f;
      }
    }
    __syncthreads();

    // Phase C: o = q @ S + attn @ u_i  (old S)
    {
      int ti = wid;
      v8f acc = {};
      for (int k0 = 0; k0 < HD; k0 += 32) {
        v16h a = lds_a_frag(&q_sh[buf][(size_t)ti * 16 * HD + k0], HD, lane);
        v16h b = lds_b_fragT(S_hT + k0, HD, lane);
        acc = wmma_f16(a, b, acc);
      }
      for (int k0 = 0; k0 < CH; k0 += 32) {
        v16h a = lds_a_frag(at_h + (size_t)ti * 16 * CH + k0, CH, lane);
        v16h b = lds_b_fragT(u_hT + k0, CH, lane);
        acc = wmma_f16(a, b, acc);
      }
#pragma unroll
      for (int r = 0; r < 8; ++r) {
        int row = ti * 16 + hi * 8 + r;
        of[cb + (size_t)row * HD + ec + col] = (_Float16)acc[r];
      }
    }
    __syncthreads();

    // Phase D: S += k^T @ u_i  (f32 accumulate, refresh transposed mirror)
    {
      int ti = wid;
      v8f acc;
#pragma unroll
      for (int r = 0; r < 8; ++r) acc[r] = S[(ti * 16 + hi * 8 + r) * 16 + col];
      for (int k0 = 0; k0 < CH; k0 += 32) {
        v16h a = lds_a_fragT(&k_sh[buf][(size_t)k0 * HD + ti * 16], HD, lane);
        v16h b = lds_b_fragT(u_hT + k0, CH, lane);
        acc = wmma_f16(a, b, acc);
      }
#pragma unroll
      for (int r = 0; r < 8; ++r) {
        int drow = ti * 16 + hi * 8 + r;
        S[drow * 16 + col]   = acc[r];
        S_hT[col * HD + drow] = (_Float16)acc[r];
      }
    }
    async_wait0();      // next-chunk staging complete before buffers swap
    __syncthreads();
    buf ^= 1;
  }
}

// =====================================================================
// Kernel 5: output projection out = o @ Wo^T + bo (f32 out). A tile
// (16x32 f16, gathered from [B,H,L,HD]) async-staged + double buffered;
// each wave owns 4 N-tiles.
// =====================================================================
__global__ void out_proj_kernel(const _Float16* __restrict__ of, const float* __restrict__ Wo,
                                const float* __restrict__ bo, float* __restrict__ out) {
  __shared__ __align__(16) _Float16 os[2][16 * 32];
  int t = threadIdx.x, lane = t & 31, wid = t >> 5;
  int mt  = blockIdx.x;                // 16-row tile (within one batch: 16 | 4096)
  int ntb = blockIdx.y;                // 256-col block
  int b_ = mt >> 8;
  int l0 = (mt & 255) * 16;

  auto stage = [&](int buf, int k0) {
    if (t < 64) {                      // 64 thr x 16B = 1KB tile
      int row = t >> 2, cb = (t & 3) * 8;
      int h = k0 >> 6;
      const _Float16* src =
          of + (((size_t)b_ * NH + h) * L_SZ + l0 + row) * HD + (k0 & 63) + cb;
      async_copy16(src, &os[buf][row * 32 + cb]);
    }
  };

  v8f acc[4] = {};
  stage(0, 0);
  int buf = 0;
  for (int k0 = 0; k0 < DMODEL; k0 += 32) {
    bool more = (k0 + 32) < DMODEL;
    if (more) { stage(buf ^ 1, k0 + 32); async_wait1(); } else { async_wait0(); }
    __syncthreads();
    v16h a = lds_a_frag(os[buf], 32, lane);
    const float* wb = Wo + (size_t)(ntb * 256 + wid * 64) * DMODEL + k0;
    __builtin_prefetch(wb + (size_t)(lane & 15) * DMODEL + 128, 0, 1);
#pragma unroll
    for (int s3 = 0; s3 < 4; ++s3) {
      v16h b = glb_b_fragT_f32(wb + (size_t)s3 * 16 * DMODEL, DMODEL, lane);
      acc[s3] = wmma_f16(a, b, acc[s3]);
    }
    __syncthreads();
    buf ^= 1;
  }
  int col = lane & 15, hi = lane >> 4;
#pragma unroll
  for (int s3 = 0; s3 < 4; ++s3) {
    int n = ntb * 256 + wid * 64 + s3 * 16 + col;
    float bi = bo[n];
#pragma unroll
    for (int r = 0; r < 8; ++r) {
      int m = mt * 16 + hi * 8 + r;
      out[(size_t)m * DMODEL + n] = acc[s3][r] + bi;
    }
  }
}

// =====================================================================
extern "C" void kernel_launch(void* const* d_in, const int* in_sizes, int n_in,
                              void* d_out, int out_size, void* d_ws, size_t ws_size,
                              hipStream_t stream) {
  const float* x  = (const float*)d_in[0];
  const float* Wq = (const float*)d_in[1];
  const float* bq = (const float*)d_in[2];
  const float* Wk = (const float*)d_in[3];
  const float* bk = (const float*)d_in[4];
  const float* Wv = (const float*)d_in[5];
  const float* bv = (const float*)d_in[6];
  const float* Wb = (const float*)d_in[7];
  const float* bb = (const float*)d_in[8];
  const float* Wo = (const float*)d_in[9];
  const float* bo = (const float*)d_in[10];
  float* out = (float*)d_out;

  const size_t MLD = (size_t)B_SZ * L_SZ * DMODEL;
  _Float16* qf = (_Float16*)d_ws;
  _Float16* kf = qf + MLD;
  _Float16* vf = kf + MLD;
  _Float16* uf = vf + MLD;
  _Float16* wf = uf + MLD;
  _Float16* of = wf + MLD;
  float* betaf = (float*)(of + MLD);

  beta_kernel<<<(B_SZ * L_SZ * NH + 255) / 256, 256, 0, stream>>>(x, Wb, bb, betaf);

  dim3 gp((B_SZ * L_SZ) / 16, DMODEL / 256, 3);
  proj_qkv_kernel<<<gp, 128, 0, stream>>>(x, Wq, bq, Wk, bk, Wv, bv, qf, kf, vf);

  chunk_prep_kernel<<<B_SZ * NH * NCHUNK, 256, 0, stream>>>(kf, vf, betaf, uf, wf);

  scan_kernel<<<B_SZ * NH * 4, 128, 0, stream>>>(qf, kf, uf, wf, of);

  dim3 go((B_SZ * L_SZ) / 16, DMODEL / 256);
  out_proj_kernel<<<go, 128, 0, stream>>>(of, Wo, bo, out);
}
